// OlmoeAttention_15272903704712
// MI455X (gfx1250) — matrix operand order repaired and checked by reading, not compiled, not run on hardware
//
#include <hip/hip_runtime.h>
#include <hip/hip_bf16.h>

#define HIDN 2048
#define NH   16
#define NKV  4
#define HD   128
#define KVD  (NKV * HD)   // 512
#define BATCH 2
#define SEQ  2048
#define EPSV 1e-5f
#define THETA 10000.0f
#define CLIPV 8.0f

typedef __attribute__((ext_vector_type(16))) __bf16 v16bf;
typedef __attribute__((ext_vector_type(8)))  __bf16 v8bf;
typedef __attribute__((ext_vector_type(8)))  float  v8f;

// ---------------------------------------------------------------- CDNA5 prims

__device__ inline v8f wmma_bf16(v16bf a, v16bf b, v8f c) {
  return __builtin_amdgcn_wmma_f32_16x16x32_bf16(
      /*neg_a=*/false, a, /*neg_b=*/false, b,
      /*c_mod=*/(short)0, c, /*reuse_a=*/false, /*reuse_b=*/false);
}

// Async global->LDS copy, 16B per lane, tracked by ASYNCcnt (gfx1250).
__device__ inline void cp_async16(unsigned int lds_off, const void* gptr) {
  asm volatile("global_load_async_to_lds_b128 %0, %1, off"
               :: "v"(lds_off), "v"((unsigned long long)gptr)
               : "memory");
}
__device__ inline void async_wait0() {
  asm volatile("s_wait_asynccnt 0x0" ::: "memory");
}
__device__ inline unsigned int lds_off32(const void* p) {
  return (unsigned int)(unsigned long long)p;  // low 32 bits = LDS byte offset
}

// A fragment 16x32 bf16. LDS tile laid out [row][k] with stride ld (elements).
__device__ inline v16bf load_a_frag(const __bf16* lds, int ld) {
  int lane = threadIdx.x & 31;
  int row  = lane & 15;
  int kh   = (lane >> 4) * 8;
  union { v16bf v; v8bf h[2]; } u;
  u.h[0] = *(const v8bf*)(lds + row * ld + kh);
  u.h[1] = *(const v8bf*)(lds + row * ld + kh + 16);
  return u.v;
}

// B fragment 32x16 bf16 from a tile laid out [n][k] (k contiguous).
__device__ inline v16bf load_b_frag(const __bf16* lds, int ld) {
  int lane = threadIdx.x & 31;
  int n    = lane & 15;
  int k0   = (lane >> 4) * 8;
  union { v16bf v; v8bf h[2]; } u;
  const __bf16* p = lds + n * ld + k0;
  u.h[0] = *(const v8bf*)p;
  u.h[1] = *(const v8bf*)(p + 16);
  return u.v;
}

__device__ inline v8bf cvt8(float4 a, float4 b) {
  v8bf v;
  v[0] = (__bf16)a.x; v[1] = (__bf16)a.y; v[2] = (__bf16)a.z; v[3] = (__bf16)a.w;
  v[4] = (__bf16)b.x; v[5] = (__bf16)b.y; v[6] = (__bf16)b.z; v[7] = (__bf16)b.w;
  return v;
}

__device__ inline float clampv(float x) {
  return fminf(fmaxf(x, -CLIPV), CLIPV);
}

// ---------------------------------------------------------------- convert

__global__ __launch_bounds__(256) void cvt_f32_bf16(
    const float* __restrict__ in, __bf16* __restrict__ out, int n8) {
  int i = blockIdx.x * 256 + threadIdx.x;
  if (i < n8) {
    const float4* p = (const float4*)(in + (size_t)i * 8);
    float4 f0 = p[0], f1 = p[1];
    *(v8bf*)(out + (size_t)i * 8) = cvt8(f0, f1);
  }
}

// ---------------------------------------------------------------- GEMM
// C[M,N] = A_bf16[M,K] @ W_bf16[N,K]^T  (row-major, K contiguous on both)
// Block tile 128x128, BK=64, 8 waves each computing 32x64. Double-buffered
// LDS filled by async global->LDS copies overlapping the WMMA compute.

#define BM 128
#define BN 128
#define BK 64
#define LDT 72   // BK + 8 pad (multiple of 8 -> 16B-aligned ds_load_b128)

__global__ __launch_bounds__(256) void gemm_bf16(
    const __bf16* __restrict__ A, const __bf16* __restrict__ W,
    float* __restrict__ C, int N, int K) {
  __shared__ __align__(16) __bf16 As[2][BM * LDT];
  __shared__ __align__(16) __bf16 Bs[2][BN * LDT];
  const int bm   = blockIdx.x * BM;
  const int bn   = blockIdx.y * BN;
  const int tid  = threadIdx.x;
  const int wave = tid >> 5;
  const int lane = tid & 31;
  const int wm   = (wave & 3) * 32;   // 0,32,64,96
  const int wn   = (wave >> 2) * 64;  // 0,64

  v8f acc[2][4];
#pragma unroll
  for (int i = 0; i < 2; ++i)
#pragma unroll
    for (int j = 0; j < 4; ++j)
#pragma unroll
      for (int e = 0; e < 8; ++e) acc[i][j][e] = 0.0f;

  // 1024 16B-chunks per matrix per stage -> 4 per thread each
  auto stage = [&](int buf, int k0) {
#pragma unroll
    for (int p = 0; p < 4; ++p) {
      int c  = tid * 4 + p;      // 0..1023
      int r  = c >> 3;           // row 0..127
      int cc = (c & 7) * 8;      // col chunk 0,8,..,56
      cp_async16(lds_off32(&As[buf][r * LDT + cc]),
                 A + (size_t)(bm + r) * K + k0 + cc);
      cp_async16(lds_off32(&Bs[buf][r * LDT + cc]),
                 W + (size_t)(bn + r) * K + k0 + cc);
    }
  };

  stage(0, 0);
  int cur = 0;
  for (int k0 = 0; k0 < K; k0 += BK) {
    async_wait0();
    __syncthreads();                       // buf[cur] visible to all waves
    if (k0 + BK < K) stage(cur ^ 1, k0 + BK);
#pragma unroll
    for (int kh = 0; kh < BK; kh += 32) {
      v16bf a0 = load_a_frag(&As[cur][(wm + 0)  * LDT + kh], LDT);
      v16bf a1 = load_a_frag(&As[cur][(wm + 16) * LDT + kh], LDT);
#pragma unroll
      for (int j = 0; j < 4; ++j) {
        v16bf b = load_b_frag(&Bs[cur][(wn + j * 16) * LDT + kh], LDT);
        acc[0][j] = wmma_bf16(a0, b, acc[0][j]);
        acc[1][j] = wmma_bf16(a1, b, acc[1][j]);
      }
    }
    __syncthreads();                       // everyone done reading buf[cur]
    cur ^= 1;
  }

  const int hi = lane >> 4;
  const int nl = lane & 15;
#pragma unroll
  for (int i = 0; i < 2; ++i)
#pragma unroll
    for (int j = 0; j < 4; ++j)
#pragma unroll
      for (int r = 0; r < 8; ++r) {
        int m = bm + wm + i * 16 + r + hi * 8;
        int n = bn + wn + j * 16 + nl;
        C[(size_t)m * N + n] = acc[i][j][r];
      }
}

// ---------------------------------------------------------------- norm+rope

__device__ inline float blk_reduce_sum(float v, float* red) {
  __syncthreads();
#pragma unroll
  for (int off = 16; off > 0; off >>= 1) v += __shfl_xor(v, off, 32);
  int wave = threadIdx.x >> 5;
  int lane = threadIdx.x & 31;
  if (lane == 0) red[wave] = v;
  __syncthreads();
  if (threadIdx.x < 32) {
    float t = (threadIdx.x < 8) ? red[threadIdx.x] : 0.0f;
#pragma unroll
    for (int off = 4; off > 0; off >>= 1) t += __shfl_xor(t, off, 32);
    if (threadIdx.x == 0) red[0] = t;
  }
  __syncthreads();
  return red[0];
}

// One block per (b,s): RMSNorm(q,k) + clip + RoPE + transpose to (b,h,s,d) bf16.
__global__ __launch_bounds__(256) void fused_norm_rope(
    const float* __restrict__ qraw, const float* __restrict__ kraw,
    const float* __restrict__ vraw, const float* __restrict__ qw,
    const float* __restrict__ kw, __bf16* __restrict__ qout,
    __bf16* __restrict__ kout, __bf16* __restrict__ vout) {
  __shared__ float red[8];
  __shared__ float qrow[HIDN];
  __shared__ float krow[KVD];
  const int bs = blockIdx.x;
  const int b  = bs / SEQ;
  const int s  = bs % SEQ;
  const int tid = threadIdx.x;

  const float* qr = qraw + (size_t)bs * HIDN;
  float ssq = 0.0f;
  for (int i = tid; i < HIDN; i += 256) { float x = qr[i]; qrow[i] = x; ssq += x * x; }
  ssq = blk_reduce_sum(ssq, red);
  const float qinv_row = rsqrtf(ssq / (float)HIDN + EPSV);

  const float* kr = kraw + (size_t)bs * KVD;
  float ssk = 0.0f;
  for (int i = tid; i < KVD; i += 256) { float x = kr[i]; krow[i] = x; ssk += x * x; }
  ssk = blk_reduce_sum(ssk, red);
  const float kinv_row = rsqrtf(ssk / (float)KVD + EPSV);

  const float t = (float)s;
  for (int p = tid; p < NH * (HD / 2); p += 256) {
    int h = p >> 6;
    int i = p & 63;
    float inv = __powf(THETA, -(float)(2 * i) / (float)HD);
    float sn, cs;
    __sincosf(t * inv, &sn, &cs);
    float x1 = clampv(qrow[h * HD + i]      * qinv_row * qw[h * HD + i]);
    float x2 = clampv(qrow[h * HD + 64 + i] * qinv_row * qw[h * HD + 64 + i]);
    size_t base = ((size_t)(b * NH + h) * SEQ + s) * HD;
    qout[base + i]      = (__bf16)(x1 * cs - x2 * sn);
    qout[base + 64 + i] = (__bf16)(x2 * cs + x1 * sn);
  }
  for (int p = tid; p < NKV * (HD / 2); p += 256) {
    int h = p >> 6;
    int i = p & 63;
    float inv = __powf(THETA, -(float)(2 * i) / (float)HD);
    float sn, cs;
    __sincosf(t * inv, &sn, &cs);
    float x1 = clampv(krow[h * HD + i]      * kinv_row * kw[h * HD + i]);
    float x2 = clampv(krow[h * HD + 64 + i] * kinv_row * kw[h * HD + 64 + i]);
    size_t base = ((size_t)(b * NKV + h) * SEQ + s) * HD;
    kout[base + i]      = (__bf16)(x1 * cs - x2 * sn);
    kout[base + 64 + i] = (__bf16)(x2 * cs + x1 * sn);
  }
  const float* vr = vraw + (size_t)bs * KVD;
  for (int p = tid; p < KVD; p += 256) {
    int h = p >> 7;
    int d = p & 127;
    size_t base = ((size_t)(b * NKV + h) * SEQ + s) * HD;
    vout[base + d] = (__bf16)clampv(vr[p]);
  }
}

// ---------------------------------------------------------------- attention
// Flash-style: block = (64 q rows, one head, one batch), 4 waves x 16 rows.
// Q/K tiles staged by async global->LDS copies; V staged transposed so its
// WMMA B-fragments are vectorized conflict-free ds_load_b128s.

#define QT 64
#define KT 64
#define QLD 136   // HD + 8
#define PLD 72    // KT + 8

__global__ __launch_bounds__(128) void attn_fwd(
    const __bf16* __restrict__ q, const __bf16* __restrict__ k,
    const __bf16* __restrict__ v, __bf16* __restrict__ out) {
  __shared__ __align__(16) __bf16 Qs[QT * QLD];
  __shared__ __align__(16) __bf16 Ks[KT * QLD];
  __shared__ __align__(16) __bf16 Vt[HD * PLD];   // [d][key], transposed
  __shared__ __align__(16) __bf16 Ps[4][16 * PLD];

  const int qb   = blockIdx.x * QT;
  const int h    = blockIdx.y;
  const int b    = blockIdx.z;
  const int kvh  = h / (NH / NKV);
  const int tid  = threadIdx.x;
  const int wave = tid >> 5;
  const int lane = tid & 31;
  const int hi   = lane >> 4;
  const int nl   = lane & 15;
  const int wm   = wave * 16;

  // stage Q tile (64 x 128 bf16) asynchronously
  const __bf16* qg = q + ((size_t)(b * NH + h) * SEQ + qb) * HD;
  for (int idx = tid; idx < QT * 16; idx += 128) {
    int r = idx >> 4;
    int c = (idx & 15) * 8;
    cp_async16(lds_off32(&Qs[r * QLD + c]), &qg[(size_t)r * HD + c]);
  }

  v8f o[8];
#pragma unroll
  for (int dt = 0; dt < 8; ++dt)
#pragma unroll
    for (int e = 0; e < 8; ++e) o[dt][e] = 0.0f;
  float mst[8], lst[8];
#pragma unroll
  for (int r = 0; r < 8; ++r) { mst[r] = -3.0e38f; lst[r] = 0.0f; }

  const __bf16* kg = k + ((size_t)(b * NKV + kvh) * SEQ) * HD;
  const __bf16* vg = v + ((size_t)(b * NKV + kvh) * SEQ) * HD;
  const float scale = 0.08838834764831845f;  // 1/sqrt(128)

  for (int kb = 0; kb < qb + QT; kb += KT) {
    __syncthreads();   // previous iteration done reading Ks/Vt
    // K tile: async copy, stays [key][d]
    for (int idx = tid; idx < KT * 16; idx += 128) {
      int r = idx >> 4;
      int c = (idx & 15) * 8;
      cp_async16(lds_off32(&Ks[r * QLD + c]), &kg[(size_t)(kb + r) * HD + c]);
    }
    // V tile: load + transposed store -> Vt[d][key]
    for (int idx = tid; idx < KT * 16; idx += 128) {
      int r = idx >> 4;
      int c = (idx & 15) * 8;
      v8bf vv = *(const v8bf*)&vg[(size_t)(kb + r) * HD + c];
#pragma unroll
      for (int j = 0; j < 8; ++j) Vt[(c + j) * PLD + r] = vv[j];
    }
    async_wait0();
    __syncthreads();   // all waves' tiles (and Q on first pass) visible

    // scores S = Q(16x128) . K^T -> 4 tiles of 16 keys
    v8f sc[4];
#pragma unroll
    for (int nt = 0; nt < 4; ++nt)
#pragma unroll
      for (int e = 0; e < 8; ++e) sc[nt][e] = 0.0f;
#pragma unroll
    for (int dk = 0; dk < HD; dk += 32) {
      v16bf aq = load_a_frag(&Qs[wm * QLD + dk], QLD);
#pragma unroll
      for (int nt = 0; nt < 4; ++nt) {
        v16bf bk = load_b_frag(&Ks[(nt * 16) * QLD + dk], QLD);
        sc[nt] = wmma_bf16(aq, bk, sc[nt]);
      }
    }

    // online softmax (each half-wave owns 8 rows; lanes nl=0..15 hold keys)
#pragma unroll
    for (int r = 0; r < 8; ++r) {
      int qi = qb + wm + r + hi * 8;
      float rowmax = -3.0e38f;
#pragma unroll
      for (int nt = 0; nt < 4; ++nt) {
        float x = sc[nt][r] * scale;
        int ki = kb + nt * 16 + nl;
        if (ki > qi) x = -1.0e9f;
        sc[nt][r] = x;
        rowmax = fmaxf(rowmax, x);
      }
#pragma unroll
      for (int off = 1; off < 16; off <<= 1)
        rowmax = fmaxf(rowmax, __shfl_xor(rowmax, off, 32));
      float mnew = fmaxf(mst[r], rowmax);
      float corr = __expf(mst[r] - mnew);
      mst[r] = mnew;
      float rsum = 0.0f;
#pragma unroll
      for (int nt = 0; nt < 4; ++nt) {
        float p = __expf(sc[nt][r] - mnew);
        sc[nt][r] = p;
        rsum += p;
      }
#pragma unroll
      for (int off = 1; off < 16; off <<= 1) rsum += __shfl_xor(rsum, off, 32);
      lst[r] = lst[r] * corr + rsum;
#pragma unroll
      for (int dt = 0; dt < 8; ++dt) o[dt][r] *= corr;
    }

    // transpose P through per-wave LDS into A-fragment layout
    __bf16* pp = &Ps[wave][0];
#pragma unroll
    for (int nt = 0; nt < 4; ++nt)
#pragma unroll
      for (int r = 0; r < 8; ++r)
        pp[(r + hi * 8) * PLD + nt * 16 + nl] = (__bf16)sc[nt][r];
    asm volatile("s_wait_dscnt 0x0" ::: "memory");  // wave-internal LDS RAW

    // O += P(16x64) . V(64x128)
#pragma unroll
    for (int kk = 0; kk < KT; kk += 32) {
      v16bf ap = load_a_frag(pp + kk, PLD);
#pragma unroll
      for (int dt = 0; dt < 8; ++dt) {
        v16bf bv = load_b_frag(&Vt[(dt * 16) * PLD + kk], PLD);
        o[dt] = wmma_bf16(ap, bv, o[dt]);
      }
    }
  }

  // epilogue: normalize and store to (b, s, HID) bf16
#pragma unroll
  for (int dt = 0; dt < 8; ++dt)
#pragma unroll
    for (int r = 0; r < 8; ++r) {
      int m = qb + wm + r + hi * 8;
      float val = o[dt][r] / lst[r];
      out[((size_t)(b * SEQ + m)) * HIDN + h * HD + dt * 16 + nl] = (__bf16)val;
    }
}

// ---------------------------------------------------------------- launch

extern "C" void kernel_launch(void* const* d_in, const int* in_sizes, int n_in,
                              void* d_out, int out_size, void* d_ws, size_t ws_size,
                              hipStream_t stream) {
  (void)in_sizes; (void)n_in; (void)out_size; (void)ws_size;
  const float* hs = (const float*)d_in[0];
  const float* Wq = (const float*)d_in[1];
  const float* Wk = (const float*)d_in[2];
  const float* Wv = (const float*)d_in[3];
  const float* Wo = (const float*)d_in[4];
  const float* qw = (const float*)d_in[5];
  const float* kw = (const float*)d_in[6];
  // d_in[7] attention_mask, d_in[8] position_ids: causal/arange computed in-kernel

  const size_t M = (size_t)BATCH * SEQ;  // 4096
  char* ws = (char*)d_ws;
  size_t off = 0;
  auto carve = [&](size_t bytes) {
    void* p = ws + off;
    off += (bytes + 255) & ~(size_t)255;
    return p;
  };
  __bf16* hsb  = (__bf16*)carve(M * HIDN * 2);
  __bf16* wqb  = (__bf16*)carve((size_t)HIDN * HIDN * 2);
  __bf16* wkb  = (__bf16*)carve((size_t)KVD  * HIDN * 2);
  __bf16* wvb  = (__bf16*)carve((size_t)KVD  * HIDN * 2);
  __bf16* wob  = (__bf16*)carve((size_t)HIDN * HIDN * 2);
  float*  qraw = (float*) carve(M * HIDN * 4);
  float*  kraw = (float*) carve(M * KVD  * 4);
  float*  vraw = (float*) carve(M * KVD  * 4);
  __bf16* qb16 = (__bf16*)carve(M * HIDN * 2);
  __bf16* kb16 = (__bf16*)carve(M * KVD  * 2);
  __bf16* vb16 = (__bf16*)carve(M * KVD  * 2);
  __bf16* ab16 = (__bf16*)carve(M * HIDN * 2);

  auto cvt = [&](const float* src, __bf16* dst, size_t n) {
    int n8 = (int)(n / 8);
    cvt_f32_bf16<<<(n8 + 255) / 256, 256, 0, stream>>>(src, dst, n8);
  };
  cvt(hs, hsb, M * HIDN);
  cvt(Wq, wqb, (size_t)HIDN * HIDN);
  cvt(Wk, wkb, (size_t)KVD * HIDN);
  cvt(Wv, wvb, (size_t)KVD * HIDN);
  cvt(Wo, wob, (size_t)HIDN * HIDN);

  gemm_bf16<<<dim3(M / BM, HIDN / BN), 256, 0, stream>>>(hsb, wqb, qraw, HIDN, HIDN);
  gemm_bf16<<<dim3(M / BM, KVD  / BN), 256, 0, stream>>>(hsb, wkb, kraw, KVD,  HIDN);
  gemm_bf16<<<dim3(M / BM, KVD  / BN), 256, 0, stream>>>(hsb, wvb, vraw, KVD,  HIDN);

  fused_norm_rope<<<dim3(BATCH * SEQ), 256, 0, stream>>>(
      qraw, kraw, vraw, qw, kw, qb16, kb16, vb16);

  attn_fwd<<<dim3(SEQ / QT, NH, BATCH), 128, 0, stream>>>(qb16, kb16, vb16, ab16);

  gemm_bf16<<<dim3(M / BM, HIDN / BN), 256, 0, stream>>>(
      ab16, wob, (float*)d_out, HIDN, HIDN);
}
